// CAAN_65068754534988
// MI455X (gfx1250) — matrix operand order repaired and checked by reading, not compiled, not run on hardware
//
#include <hip/hip_runtime.h>
#include <stddef.h>
#include <stdint.h>

typedef __bf16 bh;
typedef __attribute__((ext_vector_type(16))) __bf16 v16bf;
typedef __attribute__((ext_vector_type(8)))  __bf16 bf16x8;
typedef __attribute__((ext_vector_type(8)))  float  v8f;
typedef __attribute__((ext_vector_type(4)))  float  f32x4;
typedef __attribute__((ext_vector_type(4)))  unsigned int u32x4;
typedef __attribute__((ext_vector_type(8)))  int    i32x8;
typedef __attribute__((ext_vector_type(4)))  int    i32x4;

#define SHUF16(a,b) __builtin_shufflevector((a),(b),0,1,2,3,4,5,6,7,8,9,10,11,12,13,14,15)

constexpr int CH = 256;       // head/hidden dim
constexpr int CN = 2048;      // sequence (assets)
constexpr int CB = 8;         // batch
constexpr int CROWS = CB * CN;

// ---------------------------------------------------------------------------
// Kernel 0: W (f32, HxH row-major) -> W^T (bf16, [n][k]) so WMMA B-fragments
// are contiguous 16-byte reads per lane.
// ---------------------------------------------------------------------------
__global__ void caan_wt_kernel(const float* __restrict__ W, bh* __restrict__ WT) {
  const int t = blockIdx.x * 256 + threadIdx.x;   // 0..65535
  const int n = t >> 8, k = t & 255;
  WT[(size_t)n * CH + k] = (bh)W[(size_t)k * CH + n];
}

// ---------------------------------------------------------------------------
// Per-wave projection of a 32-row X tile (2 WMMA row-tiles sharing each
// B-fragment) against one weight matrix, with LDS-staged coalesced stores.
// C/D layout: VGPR r -> row (r + hi*8), column = lane&15 (ISA 7.12.2).
// ---------------------------------------------------------------------------
__device__ __forceinline__ void caan_project2(
    const v16bf (&xa)[2][8], const bh* __restrict__ WT, const float* __restrict__ bias,
    bh* __restrict__ outp, bh (*stage)[264],
    int row0, int r16, int hi, int lane, int transposed, int bb, int nn0)
{
  // preload this lane's 16 bias values (one per n-tile) up front
  float bnv[16];
  #pragma unroll
  for (int nt = 0; nt < 16; ++nt) bnv[nt] = bias[nt * 16 + r16];

  for (int nt = 0; nt < 16; ++nt) {
    const int n = nt * 16 + r16;
    const float bn = bnv[nt];
    v8f acc0 = 0.0f, acc1 = 0.0f;
    #pragma unroll
    for (int r = 0; r < 8; ++r) { acc0[r] = bn; acc1[r] = bn; }

    // batch-load all 8 B-fragments for this n-tile (16 x b128), then 16 WMMAs
    const bh* wr = WT + (size_t)n * CH;
    v16bf wb[8];
    #pragma unroll
    for (int kc = 0; kc < 8; ++kc) {
      const int k1 = kc * 32 + hi * 8;           // B layout: lanes0-15 K0..7,K16..23; lanes16-31 K8..15,K24..31
      bf16x8 b0 = *(const bf16x8*)(wr + k1);
      bf16x8 b1 = *(const bf16x8*)(wr + k1 + 16);
      wb[kc] = SHUF16(b0, b1);
    }
    #pragma unroll
    for (int kc = 0; kc < 8; ++kc)
      acc0 = __builtin_amdgcn_wmma_f32_16x16x32_bf16(false, xa[0][kc], false, wb[kc],
                                                     (short)0, acc0, false, false);
    #pragma unroll
    for (int kc = 0; kc < 8; ++kc)
      acc1 = __builtin_amdgcn_wmma_f32_16x16x32_bf16(false, xa[1][kc], false, wb[kc],
                                                     (short)0, acc1, false, false);

    // stash bf16 results into per-wave LDS staging tile
    if (!transposed) {                            // layout [row 0..31][col 0..255]
      #pragma unroll
      for (int r = 0; r < 8; ++r) {
        stage[r + hi * 8][n]      = (bh)acc0[r];
        stage[16 + r + hi * 8][n] = (bh)acc1[r];
      }
    } else {                                      // layout [h 0..255][key 0..31]
      bh* st = &stage[0][0];
      #pragma unroll
      for (int r = 0; r < 8; ++r) {
        st[n * 32 + r + hi * 8]      = (bh)acc0[r];
        st[n * 32 + 16 + r + hi * 8] = (bh)acc1[r];
      }
    }
  }
  asm volatile("s_wait_dscnt 0x0" ::: "memory");  // cross-lane LDS round trip (in-wave)

  // coalesced b128 write-out (32 x 16B per lane)
  if (!transposed) {
    #pragma unroll
    for (int c = 0; c < 32; ++c) {
      const int idx = c * 32 + lane;              // 0..1023 chunks of 8 elements
      const int row = idx >> 5, off = (idx & 31) * 8;
      bf16x8 vv = *(const bf16x8*)&stage[row][off];
      *(bf16x8*)(outp + (size_t)(row0 + row) * CH + off) = vv;
    }
  } else {
    const bh* st = &stage[0][0];
    #pragma unroll
    for (int c = 0; c < 32; ++c) {
      const int idx = c * 32 + lane;
      const int h = idx >> 2, off = (idx & 3) * 8;
      bf16x8 vv = *(const bf16x8*)(st + h * 32 + off);
      *(bf16x8*)(outp + ((size_t)bb * CH + h) * CN + nn0 + off) = vv;
    }
  }
}

// ---------------------------------------------------------------------------
// Kernel 1: Q/K/V = x@W + b, bf16 outputs.  4 waves/WG, 32 rows per wave.
// ---------------------------------------------------------------------------
__global__ __launch_bounds__(128) void caan_qkv_kernel(
    const float* __restrict__ x,
    const bh* __restrict__ WTq, const float* __restrict__ bq,
    const bh* __restrict__ WTk, const float* __restrict__ bk,
    const bh* __restrict__ WTv, const float* __restrict__ bv,
    bh* __restrict__ Qg, bh* __restrict__ Kg, bh* __restrict__ VTg)
{
  __shared__ __align__(16) bh sStage[4][32][264];   // per-wave store staging

  const int wave = threadIdx.x >> 5;
  const int lane = threadIdx.x & 31;
  const int r16  = lane & 15;
  const int hi   = lane >> 4;
  const int row0 = (blockIdx.x * 4 + wave) * 32;
  const int bb   = row0 >> 11;                      // batch index (32 | 2048)
  const int nn0  = row0 & (CN - 1);

  // A-fragments of X (f32 -> bf16), two 16-row tiles.
  // A layout (16-bit 16x32): lane<16 K0..7,K16..23; lane>=16 K8..15,K24..31.
  v16bf xa[2][8];
  #pragma unroll
  for (int t = 0; t < 2; ++t) {
    const float* xr = x + (size_t)(row0 + t * 16 + r16) * CH;
    #pragma unroll
    for (int kc = 0; kc < 8; ++kc) {
      const int k1 = kc * 32 + hi * 8;
      f32x4 f0 = *(const f32x4*)(xr + k1);
      f32x4 f1 = *(const f32x4*)(xr + k1 + 4);
      f32x4 f2 = *(const f32x4*)(xr + k1 + 16);
      f32x4 f3 = *(const f32x4*)(xr + k1 + 20);
      v16bf a;
      #pragma unroll
      for (int i = 0; i < 4; ++i) {
        a[i]      = (bh)f0[i];
        a[4 + i]  = (bh)f1[i];
        a[8 + i]  = (bh)f2[i];
        a[12 + i] = (bh)f3[i];
      }
      xa[t][kc] = a;
    }
  }

  bh (*stage)[264] = &sStage[wave][0];
  caan_project2(xa, WTq, bq, Qg,  stage, row0, r16, hi, lane, 0, bb, nn0);
  caan_project2(xa, WTk, bk, Kg,  stage, row0, r16, hi, lane, 0, bb, nn0);
  caan_project2(xa, WTv, bv, VTg, stage, row0, r16, hi, lane, 1, bb, nn0);
}

// ---------------------------------------------------------------------------
// Kernel 2: fused flash attention + scoring head.
// K tile staged by the Tensor Data Mover (TENSORcnt), V^T tile staged by
// async global->LDS loads (ASYNCcnt); both double-buffered.
// Grid: CB * (CN/64) WGs, 4 waves/WG, each wave owns 16 query rows.
// ---------------------------------------------------------------------------
__global__ __launch_bounds__(128) void caan_attn_kernel(
    const bh* __restrict__ Qg, const bh* __restrict__ Kg, const bh* __restrict__ VTg,
    const float* __restrict__ Ww, const float* __restrict__ bwp,
    float* __restrict__ out)
{
  __shared__ __align__(16) bh sK[2][32][264];   // 32 keys x 256 dims (pad via TDM pad feature)
  __shared__ __align__(16) bh sVT[2][256][40];  // 256 dims x 32 keys (+pad)
  __shared__ __align__(16) bh sP[4][16][40];    // per-wave P round-trip tile

  const int wave = threadIdx.x >> 5;
  const int lane = threadIdx.x & 31;
  const int r16  = lane & 15;
  const int hi   = lane >> 4;
  const int b    = blockIdx.x >> 5;
  const int t64  = blockIdx.x & 31;
  const int qrow0 = t64 * 64 + wave * 16;       // within batch
  const int tid  = threadIdx.x;

  const bh* Qb  = Qg  + (size_t)b * CN * CH;
  const bh* Kb  = Kg  + (size_t)b * CN * CH;
  const bh* VTb = VTg + (size_t)b * CH * CN;

  // --- TDM: one 2D-tile descriptor load for the 32x256 bf16 K tile -----------
  // D# per ISA 8.3/8.4: data_size=4B; tensor 128dw x 2048 rows, stride 128dw;
  // tile 128dw x 32; LDS pad: +4 dwords after every 128 dwords (-> 264-el rows).
  auto issue_k_tdm = [&](int kb, int buf) {
    const uint64_t ga = (uint64_t)(uintptr_t)(Kb + (size_t)kb * 32 * CH);
    const uint32_t la = (uint32_t)(uintptr_t)&sK[buf][0][0];
    u32x4 g0;
    g0[0] = 1u;                                          // count=1, user descriptor
    g0[1] = la;                                          // lds_addr
    g0[2] = (uint32_t)(ga & 0xFFFFFFFFu);                // global_addr[31:0]
    g0[3] = (uint32_t)((ga >> 32) & 0x1FFFFFFu) | (2u << 30);  // addr[56:32] | type=2
    const uint32_t tdim0 = 128, tdim1 = 2048, tile0 = 128, tile1 = 32, str0 = 128;
    i32x8 g1;
    g1[0] = (int)((2u << 16) | (1u << 20) | (6u << 22) | (3u << 25)); // dsz=4B,pad,128dw,4dw
    g1[1] = (int)((tdim0 & 0xFFFFu) << 16);
    g1[2] = (int)(((tdim0 >> 16) & 0xFFFFu) | ((tdim1 & 0xFFFFu) << 16));
    g1[3] = (int)(((tdim1 >> 16) & 0xFFFFu) | (tile0 << 16));
    g1[4] = (int)tile1;                                  // tile_dim1 | tile_dim2=0
    g1[5] = (int)str0;                                   // tensor_dim0_stride[31:0]
    g1[6] = 0;
    g1[7] = 0;
    i32x4 z4 = {0, 0, 0, 0};                             // groups 2/3: 2D tensor
    i32x8 z8 = {0, 0, 0, 0, 0, 0, 0, 0};
    __builtin_amdgcn_tensor_load_to_lds(g0, g1, z4, z4, z8, 0);
  };

  // --- async: 8 x b128 per thread for the 256x32 V^T tile --------------------
  auto issue_vt_async = [&](int kb, int buf) {
    #pragma unroll
    for (int c = 0; c < 8; ++c) {
      const int idx = c * 128 + tid;            // 0..1023 (8-element chunks)
      const int vh = idx >> 2, vo = (idx & 3) * 8;
      const uint32_t lv = (uint32_t)(uintptr_t)&sVT[buf][vh][vo];
      const uint64_t gv = (uint64_t)(uintptr_t)(VTb + (size_t)vh * CN + kb * 32 + vo);
      asm volatile("global_load_async_to_lds_b128 %0, %1, off"
                   :: "v"(lv), "v"(gv) : "memory");
    }
  };

  // Q A-fragments for this wave's 16 rows (resident across the whole K loop).
  v16bf qa[8];
  {
    const bh* qr = Qb + (size_t)(qrow0 + r16) * CH;
    #pragma unroll
    for (int kc = 0; kc < 8; ++kc) {
      const int k1 = kc * 32 + hi * 8;
      bf16x8 a0 = *(const bf16x8*)(qr + k1);
      bf16x8 a1 = *(const bf16x8*)(qr + k1 + 16);
      qa[kc] = SHUF16(a0, a1);
    }
  }

  v8f acc[16];
  #pragma unroll
  for (int ht = 0; ht < 16; ++ht) acc[ht] = 0.0f;
  float m[8], l[8];
  #pragma unroll
  for (int r = 0; r < 8; ++r) { m[r] = -1e30f; l[r] = 0.f; }

  if (wave == 0) issue_k_tdm(0, 0);
  issue_vt_async(0, 0);

  for (int kb = 0; kb < CN / 32; ++kb) {
    const int buf = kb & 1;
    // prefetch next stage into the other buffer (its readers finished at the
    // previous end-of-iteration barrier), then ensure the current stage landed
    if (kb + 1 < CN / 32) {
      if (wave == 0) issue_k_tdm(kb + 1, buf ^ 1);
      issue_vt_async(kb + 1, buf ^ 1);
      if (wave == 0) __builtin_amdgcn_s_wait_tensorcnt(1);
      asm volatile("s_wait_asynccnt 0x8" ::: "memory");   // <=8 outstanding
    } else {
      if (wave == 0) __builtin_amdgcn_s_wait_tensorcnt(0);
      asm volatile("s_wait_asynccnt 0x0" ::: "memory");
    }
    __syncthreads();

    // --- S = Q K^T (16 rows x 32 keys), K-dim = 256 in 8 WMMA steps ---
    v8f s[2];
    #pragma unroll
    for (int nt = 0; nt < 2; ++nt) {
      v8f sc = 0.0f;
      const bh* krow = &sK[buf][nt * 16 + r16][0];
      #pragma unroll
      for (int kc = 0; kc < 8; ++kc) {
        const int k1 = kc * 32 + hi * 8;
        bf16x8 b0 = *(const bf16x8*)(krow + k1);
        bf16x8 b1 = *(const bf16x8*)(krow + k1 + 16);
        v16bf kbf = SHUF16(b0, b1);
        sc = __builtin_amdgcn_wmma_f32_16x16x32_bf16(false, qa[kc], false, kbf,
                                                     (short)0, sc, false, false);
      }
      s[nt] = sc;
    }

    // --- online softmax (rows in VGPR index, cols in lanes; 16-lane reductions) ---
    const float SCL = 0.0625f;                   // 1/sqrt(256)
    const float L2E = 1.44269504088896340736f;
    float alpha[8];
    #pragma unroll
    for (int r = 0; r < 8; ++r) {
      float s0 = s[0][r] * SCL, s1 = s[1][r] * SCL;
      float mx = fmaxf(s0, s1);
      mx = fmaxf(mx, __shfl_xor(mx, 1));
      mx = fmaxf(mx, __shfl_xor(mx, 2));
      mx = fmaxf(mx, __shfl_xor(mx, 4));
      mx = fmaxf(mx, __shfl_xor(mx, 8));
      const float mn = fmaxf(m[r], mx);
      const float a  = __builtin_amdgcn_exp2f((m[r] - mn) * L2E);
      const float p0 = __builtin_amdgcn_exp2f((s0 - mn) * L2E);
      const float p1 = __builtin_amdgcn_exp2f((s1 - mn) * L2E);
      s[0][r] = p0; s[1][r] = p1;
      float rs = p0 + p1;
      rs += __shfl_xor(rs, 1);
      rs += __shfl_xor(rs, 2);
      rs += __shfl_xor(rs, 4);
      rs += __shfl_xor(rs, 8);
      l[r] = l[r] * a + rs;
      m[r] = mn;
      alpha[r] = a;
    }
    #pragma unroll
    for (int ht = 0; ht < 16; ++ht)
      #pragma unroll
      for (int r = 0; r < 8; ++r)
        acc[ht][r] *= alpha[r];

    // --- P (C-layout f32) -> bf16 A-fragment via per-wave LDS round-trip ---
    #pragma unroll
    for (int nt = 0; nt < 2; ++nt)
      #pragma unroll
      for (int r = 0; r < 8; ++r)
        sP[wave][r + hi * 8][nt * 16 + r16] = (bh)s[nt][r];
    asm volatile("s_wait_dscnt 0x0" ::: "memory");
    bf16x8 p0v = *(const bf16x8*)&sP[wave][r16][hi * 8];
    bf16x8 p1v = *(const bf16x8*)&sP[wave][r16][16 + hi * 8];
    v16bf pa = SHUF16(p0v, p1v);

    // --- O += P V  (16 column tiles over H=256) ---
    #pragma unroll
    for (int ht = 0; ht < 16; ++ht) {
      const bh* vrow = &sVT[buf][ht * 16 + r16][0];
      bf16x8 v0 = *(const bf16x8*)(vrow + hi * 8);
      bf16x8 v1 = *(const bf16x8*)(vrow + hi * 8 + 16);
      v16bf vb = SHUF16(v0, v1);
      acc[ht] = __builtin_amdgcn_wmma_f32_16x16x32_bf16(false, pa, false, vb,
                                                        (short)0, acc[ht], false, false);
    }
    __syncthreads();   // readers done -> next iteration may refill this buffer
  }

  // --- final scores: (O/l) . Ww + bw ---
  float ww[16];
  #pragma unroll
  for (int ht = 0; ht < 16; ++ht) ww[ht] = Ww[ht * 16 + r16];
  const float bw = *bwp;
  #pragma unroll
  for (int r = 0; r < 8; ++r) {
    float p = 0.f;
    #pragma unroll
    for (int ht = 0; ht < 16; ++ht) p += acc[ht][r] * ww[ht];
    p += __shfl_xor(p, 1);
    p += __shfl_xor(p, 2);
    p += __shfl_xor(p, 4);
    p += __shfl_xor(p, 8);
    const float score = p / l[r] + bw;
    if (r16 == r)
      out[(size_t)b * CN + qrow0 + r + hi * 8] = score;
  }
}

// ---------------------------------------------------------------------------
extern "C" void kernel_launch(void* const* d_in, const int* in_sizes, int n_in,
                              void* d_out, int out_size, void* d_ws, size_t ws_size,
                              hipStream_t stream) {
  (void)in_sizes; (void)n_in; (void)out_size; (void)ws_size;
  const float* x  = (const float*)d_in[0];
  const float* Wq = (const float*)d_in[1];
  const float* bq = (const float*)d_in[2];
  const float* Wk = (const float*)d_in[3];
  const float* bk = (const float*)d_in[4];
  const float* Wv = (const float*)d_in[5];
  const float* bv = (const float*)d_in[6];
  const float* Ww = (const float*)d_in[7];
  const float* bw = (const float*)d_in[8];

  char* ws = (char*)d_ws;
  size_t off = 0;
  auto take = [&](size_t bytes) {
    char* p = ws + off;
    off += (bytes + 255) & ~(size_t)255;
    return p;
  };
  bh* Qg  = (bh*)take((size_t)CROWS * CH * 2);
  bh* Kg  = (bh*)take((size_t)CROWS * CH * 2);
  bh* VTg = (bh*)take((size_t)CROWS * CH * 2);
  bh* WTq = (bh*)take((size_t)CH * CH * 2);
  bh* WTk = (bh*)take((size_t)CH * CH * 2);
  bh* WTv = (bh*)take((size_t)CH * CH * 2);

  caan_wt_kernel<<<CH * CH / 256, 256, 0, stream>>>(Wq, WTq);
  caan_wt_kernel<<<CH * CH / 256, 256, 0, stream>>>(Wk, WTk);
  caan_wt_kernel<<<CH * CH / 256, 256, 0, stream>>>(Wv, WTv);

  caan_qkv_kernel<<<CROWS / 128, 128, 0, stream>>>(x, WTq, bq, WTk, bk, WTv, bv,
                                                   Qg, Kg, VTg);

  caan_attn_kernel<<<CB * (CN / 64), 128, 0, stream>>>(Qg, Kg, VTg, Ww, bw,
                                                       (float*)d_out);
}
